// ModelA_37563783971151
// MI455X (gfx1250) — matrix operand order repaired and checked by reading, not compiled, not run on hardware
//
#include <hip/hip_runtime.h>
#include <hip/hip_bf16.h>
#include <cstdint>
#include <cstddef>

// ---------- types ----------
typedef __attribute__((ext_vector_type(16))) __bf16        v16bf;
typedef __attribute__((ext_vector_type(8)))  float         v8f;
typedef __attribute__((ext_vector_type(4)))  unsigned int  u32x4;
typedef __attribute__((ext_vector_type(2)))  unsigned int  u32x2;
typedef __attribute__((ext_vector_type(4)))  float         f32x4;

union Frag { u32x4 u[2]; v16bf v; };

__device__ __forceinline__ unsigned short f2bf(float f) {
    unsigned int u = __float_as_uint(f);
    u += 0x7fffu + ((u >> 16) & 1u);       // round-to-nearest-even
    return (unsigned short)(u >> 16);
}

// ---------- problem constants ----------
#define BB   8
#define TT   8192
#define XX   64
#define HH   128
#define ROWS (BB * TT)        // 65536
#define NCH  64               // chunks per sequence
#define CLEN 128              // chunk length (NCH*CLEN == TT)

// ---------- elementwise converts ----------
__global__ __launch_bounds__(256) void k_f32_to_bf16(const float* __restrict__ src,
                                                     unsigned short* __restrict__ dst, int n) {
    int i = blockIdx.x * 256 + threadIdx.x, stride = gridDim.x * 256;
    for (; i < n; i += stride) dst[i] = f2bf(src[i]);
}

// (K x N) f32 -> (Npad x K) bf16 transposed, zero pad n >= N
__global__ __launch_bounds__(256) void k_wT_bf16(const float* __restrict__ w,
                                                 unsigned short* __restrict__ wT,
                                                 int K, int N, int Npad) {
    int i = blockIdx.x * 256 + threadIdx.x, total = K * Npad, stride = gridDim.x * 256;
    for (; i < total; i += stride) {
        int n = i / K, k = i - n * K;
        wT[i] = f2bf((n < N) ? w[k * N + n] : 0.f);
    }
}

// ---------- generic WMMA GEMM (embed / proj) ----------
// block: 256 threads = 8 waves -> 32 rows x 64 cols (wave = one 16x16 tile)
// epi: 1 = leaky_relu(+bias), 2 = softplus(+bias)
__global__ __launch_bounds__(256) void k_gemm(const unsigned short* __restrict__ A,
                                              const unsigned short* __restrict__ BT,
                                              const float* __restrict__ bias,
                                              float* __restrict__ out,
                                              int K, int Nreal, int Nstore, int ldOut, int epi) {
    const int lane = threadIdx.x & 31;
    const int wave = threadIdx.x >> 5;
    const int l15  = lane & 15;
    const int kg   = lane >> 4;
    const int rowBase = blockIdx.y * 32 + (wave >> 2) * 16;
    const int nBase   = blockIdx.x * 64 + (wave & 3) * 16;

    const unsigned short* Ap = A  + (size_t)(rowBase + l15) * K + kg * 16;
    const unsigned short* Bp = BT + (size_t)(nBase   + l15) * K + kg * 16;

    v8f acc = {};
    for (int kb = 0; kb < K; kb += 32) {
        Frag fa, fb;
        fa.u[0] = *(const u32x4*)(Ap + kb);        // K = kb + kg*16 + 0..7  (VGPR 0..3)
        fa.u[1] = *(const u32x4*)(Ap + kb + 8);    // K = kb + kg*16 + 8..15 (VGPR 4..7)
        fb.u[0] = *(const u32x4*)(Bp + kb);        // K = kb + kg*16 + 0..7
        fb.u[1] = *(const u32x4*)(Bp + kb + 8);    // K = kb + kg*16 + 8..15
        acc = __builtin_amdgcn_wmma_f32_16x16x32_bf16(false, fa.v, false, fb.v,
                                                      (short)0, acc, false, false);
    }

    const int col = nBase + l15;
    const int r0  = rowBase + kg * 8;
    if (col < Nstore) {
        const float bv = (col < Nreal) ? bias[col] : 0.f;
        #pragma unroll
        for (int r = 0; r < 8; ++r) {
            float v = acc[r] + bv;
            if (epi == 1)      v = (v >= 0.f) ? v : 0.01f * v;
            else if (epi == 2) v = (v > 20.f) ? v : log1pf(expf(v));
            out[(size_t)(r0 + r) * ldOut + col] = v;
        }
    }
}

// ---------- fused GRU GEMM + gate ----------
// Each wave owns a hidden tile (cols nBase..nBase+15) AND the matching gate tile
// (cols nBase+128..), shares one A fragment, and writes a = 1-z, bb = z*g(hidden)
// directly — the (ROWS x 256) hg tensor never touches memory.
__global__ __launch_bounds__(256) void k_gemm_gate(const unsigned short* __restrict__ A,
                                                   const unsigned short* __restrict__ BT,
                                                   const float* __restrict__ bias,
                                                   float* __restrict__ a_out,
                                                   float* __restrict__ bb_out) {
    const int lane = threadIdx.x & 31;
    const int wave = threadIdx.x >> 5;
    const int l15  = lane & 15;
    const int kg   = lane >> 4;
    const int rowBase = blockIdx.y * 32 + (wave >> 2) * 16;
    const int nBase   = blockIdx.x * 64 + (wave & 3) * 16;       // hidden tile (0..127)

    const unsigned short* Ap = A  + (size_t)(rowBase + l15) * HH + kg * 16;
    const unsigned short* Bh = BT + (size_t)(nBase + l15) * HH + kg * 16;        // hidden cols
    const unsigned short* Bg = BT + (size_t)(nBase + 128 + l15) * HH + kg * 16;  // gate cols

    v8f acch = {}, accg = {};
    #pragma unroll
    for (int kb = 0; kb < HH; kb += 32) {
        Frag fa, fbh, fbg;
        fa.u[0]  = *(const u32x4*)(Ap + kb);
        fa.u[1]  = *(const u32x4*)(Ap + kb + 8);
        fbh.u[0] = *(const u32x4*)(Bh + kb);
        fbh.u[1] = *(const u32x4*)(Bh + kb + 8);
        fbg.u[0] = *(const u32x4*)(Bg + kb);
        fbg.u[1] = *(const u32x4*)(Bg + kb + 8);
        acch = __builtin_amdgcn_wmma_f32_16x16x32_bf16(false, fa.v, false, fbh.v,
                                                       (short)0, acch, false, false);
        accg = __builtin_amdgcn_wmma_f32_16x16x32_bf16(false, fa.v, false, fbg.v,
                                                       (short)0, accg, false, false);
    }

    const int col = nBase + l15;
    const int r0  = rowBase + kg * 8;
    const float bh = bias[col];
    const float bg = bias[col + 128];
    #pragma unroll
    for (int r = 0; r < 8; ++r) {
        float hid = acch[r] + bh;
        float gat = accg[r] + bg;
        float z   = 1.f / (1.f + expf(-gat));
        float gv  = (hid >= 0.f) ? hid + 0.5f : 1.f / (1.f + expf(-hid));
        size_t o = (size_t)(r0 + r) * HH + col;
        a_out[o]  = 1.f - z;
        bb_out[o] = z * gv;
    }
}

// ---------- blocked scan over T ----------
// pass1: per (b, chunk j, channel c): local scan (h0=0), store local h + chunk aggregate (A, B)
__global__ __launch_bounds__(256) void k_scan1(const float* __restrict__ a,
                                               const float* __restrict__ bb,
                                               float* __restrict__ hout,
                                               float* __restrict__ aggA,
                                               float* __restrict__ aggB) {
    int gid = blockIdx.x * 256 + threadIdx.x;          // 65536 threads
    int c = gid & 127, j = (gid >> 7) & 63, b = gid >> 13;
    size_t base = ((size_t)b * TT + (size_t)j * CLEN) * HH + c;
    float h = 0.f, A = 1.f;
    for (int t = 0; t < CLEN; ++t) {
        size_t idx = base + (size_t)t * HH;
        float at = a[idx];
        h = fmaf(at, h, bb[idx]);
        A *= at;
        hout[idx] = h;
    }
    int ai = (b * NCH + j) * HH + c;
    aggA[ai] = A; aggB[ai] = h;
}

// pass2: per (b, c): exclusive prefix over the 64 chunk aggregates -> carry
__global__ __launch_bounds__(256) void k_scan2(const float* __restrict__ aggA,
                                               const float* __restrict__ aggB,
                                               float* __restrict__ carry) {
    int gid = blockIdx.x * 256 + threadIdx.x;          // 1024 threads
    if (gid >= BB * HH) return;
    int c = gid & 127, b = gid >> 7;
    float h = 0.f;
    for (int j = 0; j < NCH; ++j) {
        int ai = (b * NCH + j) * HH + c;
        carry[ai] = h;
        h = fmaf(aggA[ai], h, aggB[ai]);
    }
}

// pass3: h[t] = h_local[t] + (prod_{t'<=t} a) * carry
__global__ __launch_bounds__(256) void k_scan3(const float* __restrict__ a,
                                               const float* __restrict__ carry,
                                               float* __restrict__ hio) {
    int gid = blockIdx.x * 256 + threadIdx.x;
    int c = gid & 127, j = (gid >> 7) & 63, b = gid >> 13;
    float c0 = carry[(b * NCH + j) * HH + c];
    if (c0 == 0.f) return;                              // chunk 0 always skips
    size_t base = ((size_t)b * TT + (size_t)j * CLEN) * HH + c;
    float P = 1.f;
    for (int t = 0; t < CLEN; ++t) {
        size_t idx = base + (size_t)t * HH;
        P *= a[idx];
        hio[idx] += P * c0;
    }
}

// ---------- layernorm over H=128, f32 in -> bf16 out; one wave32 per row ----------
__global__ __launch_bounds__(256) void k_ln(const float* __restrict__ h,
                                            const float* __restrict__ gamma,
                                            const float* __restrict__ beta,
                                            unsigned short* __restrict__ outbf, int rows) {
    int wave = threadIdx.x >> 5, lane = threadIdx.x & 31;
    int row = blockIdx.x * 8 + wave;
    if (row >= rows) return;
    int c = lane * 4;
    f32x4 v = *(const f32x4*)(h + (size_t)row * HH + c);
    float s = v[0] + v[1] + v[2] + v[3];
    float q = v[0]*v[0] + v[1]*v[1] + v[2]*v[2] + v[3]*v[3];
    for (int m = 16; m >= 1; m >>= 1) { s += __shfl_xor(s, m, 32); q += __shfl_xor(q, m, 32); }
    float mean = s * (1.f / HH);
    float var  = q * (1.f / HH) - mean * mean;
    float rs   = rsqrtf(var + 1e-5f);
    f32x4 g  = *(const f32x4*)(gamma + c);
    f32x4 bt = *(const f32x4*)(beta + c);
    unsigned int o0 = f2bf((v[0] - mean) * rs * g[0] + bt[0]);
    unsigned int o1 = f2bf((v[1] - mean) * rs * g[1] + bt[1]);
    unsigned int o2 = f2bf((v[2] - mean) * rs * g[2] + bt[2]);
    unsigned int o3 = f2bf((v[3] - mean) * rs * g[3] + bt[3]);
    u32x2 pk; pk[0] = o0 | (o1 << 16); pk[1] = o2 | (o3 << 16);
    *(u32x2*)(outbf + (size_t)row * HH + c) = pk;
}

// ---------- host launch ----------
extern "C" void kernel_launch(void* const* d_in, const int* in_sizes, int n_in,
                              void* d_out, int out_size, void* d_ws, size_t ws_size,
                              hipStream_t stream) {
    // setup_inputs() dict order:
    const float* x       = (const float*)d_in[0];
    const float* embed_w = (const float*)d_in[1];
    const float* embed_b = (const float*)d_in[2];
    const float* ln0_g   = (const float*)d_in[3];
    const float* ln0_b   = (const float*)d_in[4];
    const float* proj_w  = (const float*)d_in[5];
    const float* proj_b  = (const float*)d_in[6];
    const float* gru_w[3] = { (const float*)d_in[7],  (const float*)d_in[11], (const float*)d_in[15] };
    const float* gru_b[3] = { (const float*)d_in[8],  (const float*)d_in[12], (const float*)d_in[16] };
    const float* ln_g[3]  = { (const float*)d_in[9],  (const float*)d_in[13], (const float*)d_in[17] };
    const float* ln_b[3]  = { (const float*)d_in[10], (const float*)d_in[14], (const float*)d_in[18] };

    // workspace carve-up (256B aligned)
    char* w = (char*)d_ws;
    size_t off = 0;
    auto alloc = [&](size_t bytes) -> void* {
        void* p = w + off;
        off += (bytes + 255) & ~(size_t)255;
        return p;
    };
    unsigned short* x_bf  = (unsigned short*)alloc((size_t)ROWS * XX * 2);
    unsigned short* h_bf  = (unsigned short*)alloc((size_t)ROWS * HH * 2);
    float* a_buf = (float*)alloc((size_t)ROWS * HH * 4);
    float* b_buf = (float*)alloc((size_t)ROWS * HH * 4);
    float* h_f32 = (float*)alloc((size_t)ROWS * HH * 4);
    float* aggA  = (float*)alloc((size_t)BB * NCH * HH * 4);
    float* aggB  = (float*)alloc((size_t)BB * NCH * HH * 4);
    float* carry = (float*)alloc((size_t)BB * NCH * HH * 4);
    unsigned short* wTe = (unsigned short*)alloc((size_t)HH * XX * 2);       // 128 x 64
    unsigned short* wTg[3];
    for (int i = 0; i < 3; ++i) wTg[i] = (unsigned short*)alloc((size_t)256 * HH * 2); // 256 x 128
    unsigned short* wTp = (unsigned short*)alloc((size_t)64 * HH * 2);       // 64 x 128 (padded)
    (void)ws_size; (void)n_in; (void)in_sizes; (void)out_size;

    // converts (graph-safe, deterministic)
    k_f32_to_bf16<<<8192, 256, 0, stream>>>(x, x_bf, ROWS * XX);
    k_wT_bf16<<<32,  256, 0, stream>>>(embed_w, wTe, XX, HH, HH);            // K=64, N=128
    for (int i = 0; i < 3; ++i)
        k_wT_bf16<<<128, 256, 0, stream>>>(gru_w[i], wTg[i], HH, 256, 256);  // K=128, N=256
    k_wT_bf16<<<32,  256, 0, stream>>>(proj_w, wTp, HH, 50, 64);             // K=128, pad to 64

    const dim3 blk(256);

    // embed: leaky_relu(x @ We + b) -> h_f32 ; then LN -> h_bf
    k_gemm<<<dim3(2, ROWS / 32), blk, 0, stream>>>(x_bf, wTe, embed_b, h_f32,
                                                   XX, HH, HH, HH, /*epi=*/1);
    k_ln<<<ROWS / 8, blk, 0, stream>>>(h_f32, ln0_g, ln0_b, h_bf, ROWS);

    // three minGRU blocks (GEMM+gate fused; hg never hits memory)
    for (int i = 0; i < 3; ++i) {
        k_gemm_gate<<<dim3(2, ROWS / 32), blk, 0, stream>>>(h_bf, wTg[i], gru_b[i],
                                                            a_buf, b_buf);
        k_scan1<<<256, blk, 0, stream>>>(a_buf, b_buf, h_f32, aggA, aggB);
        k_scan2<<<4,   blk, 0, stream>>>(aggA, aggB, carry);
        k_scan3<<<256, blk, 0, stream>>>(a_buf, carry, h_f32);
        k_ln<<<ROWS / 8, blk, 0, stream>>>(h_f32, ln_g[i], ln_b[i], h_bf, ROWS);
    }

    // projection: softplus(h @ Wp + b) -> d_out (ROWS x 50)
    k_gemm<<<dim3(1, ROWS / 32), blk, 0, stream>>>(h_bf, wTp, proj_b, (float*)d_out,
                                                   HH, 50, 50, 50, /*epi=*/2);
}
